// CrossModalMatchingNetwork_49005576847522
// MI455X (gfx1250) — matrix operand order repaired and checked by reading, not compiled, not run on hardware
//
#include <hip/hip_runtime.h>
#include <hip/hip_bf16.h>

// ---------------------------------------------------------------------------
// CDNA5 / gfx1250 CrossModalMatchingNetwork.
//   v = vis @ Wv^T + bv   [16,1024,512]   (bf16 WMMA, f32 accum)
//   t = txt @ Wt^T + bt   [16,1024,512]
//   out[b,i,j] = <v_bi,t_bj> / max(|v_bi||t_bj|, 1e-8)
// Pipeline: (1) elementwise f32->bf16 staging (native cvt, L2-resident),
// (2) bf16 WMMA projection GEMMs, (3) row norms, (4) batched bf16 WMMA dots
// + cosine normalize.  GEMM waves own 32x64 tiles: 8 v_wmma per 6 b128
// fragment loads (2 A + 4 B), blocks tile 128x128 for L0 fragment sharing.
// ---------------------------------------------------------------------------

typedef __attribute__((ext_vector_type(16))) __bf16 v16bf;
typedef __attribute__((ext_vector_type(8)))  float  v8f;
typedef __attribute__((ext_vector_type(16))) unsigned short v16us;

#define HDIM 512
#define SEQ  1024
#define EPSV 1e-8f

static __device__ __forceinline__ unsigned short f2bf(float f) {
  __bf16 h = (__bf16)f;                       // hardware RNE cvt
  return __builtin_bit_cast(unsigned short, h);
}
static __device__ __forceinline__ float bf2f(unsigned short h) {
  unsigned u = ((unsigned)h) << 16;
  return __builtin_bit_cast(float, u);
}
static __device__ __forceinline__ unsigned pack2(float lo, float hi) {
  return (unsigned)f2bf(lo) | ((unsigned)f2bf(hi) << 16);
}

// A fragment (16-bit A 16x32): lane m=lane&15, kb=(lane>>4)*8,
// holds K = 2*kb + {0..15} (contiguous, matches B): two 16B chunks.
static __device__ __forceinline__ v16bf afrag(const unsigned short* __restrict__ rowp,
                                              int k0, int kb) {
  v16us u;
  *reinterpret_cast<uint4*>(&u)       = *reinterpret_cast<const uint4*>(rowp + k0 + 2 * kb);
  *(reinterpret_cast<uint4*>(&u) + 1) = *reinterpret_cast<const uint4*>(rowp + k0 + 2 * kb + 8);
  return __builtin_bit_cast(v16bf, u);
}
// B fragment (16-bit B 32x16): lane n=lane&15 holds column n,
// K = (lane>>4)*16 + {0..15}: one 32B contiguous (32B-aligned) chunk.
static __device__ __forceinline__ v16bf bfrag(const unsigned short* __restrict__ colp,
                                              int k0, int kh16) {
  v16us u;
  const uint4* p = reinterpret_cast<const uint4*>(colp + k0 + kh16);
  *reinterpret_cast<uint4*>(&u)       = p[0];
  *(reinterpret_cast<uint4*>(&u) + 1) = p[1];
  return __builtin_bit_cast(v16bf, u);
}

static __device__ __forceinline__ v8f wmma_bf16(v16bf a, v16bf b, v8f c) {
  return __builtin_amdgcn_wmma_f32_16x16x32_bf16(
      false, a, false, b, (short)0, c, false, false);
}

// ---------------------------------------------------------------------------
// Elementwise f32 -> bf16 (8 elements/thread, b128 in, b128 out).
// ---------------------------------------------------------------------------
__global__ __launch_bounds__(256)
void cvt_kernel(const float* __restrict__ src, unsigned short* __restrict__ dst) {
  const size_t i = ((size_t)blockIdx.x * 256 + threadIdx.x) * 8;
  float4 a = *reinterpret_cast<const float4*>(src + i);
  float4 b = *reinterpret_cast<const float4*>(src + i + 4);
  uint4 o;
  o.x = pack2(a.x, a.y);
  o.y = pack2(a.z, a.w);
  o.z = pack2(b.x, b.y);
  o.w = pack2(b.z, b.w);
  *reinterpret_cast<uint4*>(dst + i) = o;
}

// ---------------------------------------------------------------------------
// Projection: Y = Xbf @ Wbf^T + bias (bf16 out).
// Block tile 128(M) x 128(N); 8 waves in 4(M) x 2(N); wave tile 32x64.
// Grid: (rows/128, 512/128).
// ---------------------------------------------------------------------------
__global__ __launch_bounds__(256)
void proj_kernel(const unsigned short* __restrict__ Xbf,
                 const unsigned short* __restrict__ Wbf,
                 const float* __restrict__ bias,
                 unsigned short* __restrict__ Y, int K) {
  const int lane  = threadIdx.x & 31;
  const int wave  = threadIdx.x >> 5;
  const int mrow0 = blockIdx.x * 128 + (wave & 3) * 32;
  const int ncol0 = blockIdx.y * 128 + (wave >> 2) * 64;
  const int m     = lane & 15;
  const int kb    = (lane >> 4) * 8;
  const int kh16  = (lane >> 4) * 16;

  const unsigned short* xrow0 = Xbf + (size_t)(mrow0 + m) * K;
  const unsigned short* xrow1 = Xbf + (size_t)(mrow0 + 16 + m) * K;
  const unsigned short* wrow[4];
#pragma unroll
  for (int t = 0; t < 4; ++t)
    wrow[t] = Wbf + (size_t)(ncol0 + t * 16 + m) * K;

  v8f acc[2][4];
#pragma unroll
  for (int i = 0; i < 2; ++i)
#pragma unroll
    for (int t = 0; t < 4; ++t) acc[i][t] = v8f{};

#pragma unroll 2
  for (int k0 = 0; k0 < K; k0 += 32) {
    v16bf a0 = afrag(xrow0, k0, kb);
    v16bf a1 = afrag(xrow1, k0, kb);
#pragma unroll
    for (int t = 0; t < 4; ++t) {
      v16bf b = bfrag(wrow[t], k0, kh16);
      acc[0][t] = wmma_bf16(a0, b, acc[0][t]);
      acc[1][t] = wmma_bf16(a1, b, acc[1][t]);
    }
  }

  const int hi = lane >> 4;
#pragma unroll
  for (int t = 0; t < 4; ++t) {
    const int n = ncol0 + t * 16 + m;
    const float bia = bias[n];
#pragma unroll
    for (int i = 0; i < 2; ++i)
#pragma unroll
      for (int r = 0; r < 8; ++r) {
        const int row = mrow0 + i * 16 + r + 8 * hi;
        Y[(size_t)row * HDIM + n] = f2bf(acc[i][t][r] + bia);
      }
  }
}

// ---------------------------------------------------------------------------
// Row L2 norms of bf16 matrix [rows x 512]: one wave per row.
// ---------------------------------------------------------------------------
__global__ __launch_bounds__(256)
void norm_kernel(const unsigned short* __restrict__ Y, float* __restrict__ nrm) {
  const int lane = threadIdx.x & 31;
  const int wave = threadIdx.x >> 5;
  const size_t row = (size_t)blockIdx.x * 8 + wave;
  const uint4* p = reinterpret_cast<const uint4*>(Y + row * HDIM + lane * 16);
  uint4 q0 = p[0];
  uint4 q1 = p[1];
  unsigned w[8] = {q0.x, q0.y, q0.z, q0.w, q1.x, q1.y, q1.z, q1.w};
  float s = 0.f;
#pragma unroll
  for (int j = 0; j < 8; ++j) {
    float a = bf2f((unsigned short)(w[j] & 0xffffu));
    float b = bf2f((unsigned short)(w[j] >> 16));
    s += a * a + b * b;
  }
#pragma unroll
  for (int mask = 16; mask >= 1; mask >>= 1)
    s += __shfl_xor(s, mask, 32);
  if (lane == 0) nrm[row] = sqrtf(s);
}

// ---------------------------------------------------------------------------
// Batched dots + cosine normalization.
// Block tile 128(M) x 128(N); 8 waves in 4(M) x 2(N); wave tile 32x64.
// Grid: (8, 8, 16).  K = 512 in 16 steps of 32.
// ---------------------------------------------------------------------------
__global__ __launch_bounds__(256)
void dots_kernel(const unsigned short* __restrict__ Vbf,
                 const unsigned short* __restrict__ Tbf,
                 const float* __restrict__ vn, const float* __restrict__ tn,
                 float* __restrict__ out) {
  const int lane  = threadIdx.x & 31;
  const int wave  = threadIdx.x >> 5;
  const int b     = blockIdx.z;
  const int mrow0 = blockIdx.y * 128 + (wave & 3) * 32;
  const int ncol0 = blockIdx.x * 128 + (wave >> 2) * 64;
  const int m     = lane & 15;
  const int kb    = (lane >> 4) * 8;
  const int kh16  = (lane >> 4) * 16;

  const unsigned short* vrow0 = Vbf + (size_t)(b * SEQ + mrow0 + m) * HDIM;
  const unsigned short* vrow1 = Vbf + (size_t)(b * SEQ + mrow0 + 16 + m) * HDIM;
  const unsigned short* trow[4];
#pragma unroll
  for (int t = 0; t < 4; ++t)
    trow[t] = Tbf + (size_t)(b * SEQ + ncol0 + t * 16 + m) * HDIM;

  v8f acc[2][4];
#pragma unroll
  for (int i = 0; i < 2; ++i)
#pragma unroll
    for (int t = 0; t < 4; ++t) acc[i][t] = v8f{};

#pragma unroll 2
  for (int k0 = 0; k0 < HDIM; k0 += 32) {
    v16bf a0 = afrag(vrow0, k0, kb);
    v16bf a1 = afrag(vrow1, k0, kb);
#pragma unroll
    for (int t = 0; t < 4; ++t) {
      v16bf bm = bfrag(trow[t], k0, kh16);
      acc[0][t] = wmma_bf16(a0, bm, acc[0][t]);
      acc[1][t] = wmma_bf16(a1, bm, acc[1][t]);
    }
  }

  const int hi = lane >> 4;
  float vnr[2][8];
#pragma unroll
  for (int i = 0; i < 2; ++i)
#pragma unroll
    for (int r = 0; r < 8; ++r)
      vnr[i][r] = vn[(size_t)b * SEQ + mrow0 + i * 16 + r + 8 * hi];

#pragma unroll
  for (int t = 0; t < 4; ++t) {
    const int n = ncol0 + t * 16 + m;
    const float tnv = tn[(size_t)b * SEQ + n];
#pragma unroll
    for (int i = 0; i < 2; ++i)
#pragma unroll
      for (int r = 0; r < 8; ++r) {
        const int row = mrow0 + i * 16 + r + 8 * hi;
        const float denom = fmaxf(vnr[i][r] * tnv, EPSV);
        out[((size_t)b * SEQ + row) * SEQ + n] = acc[i][t][r] / denom;
      }
  }
}

// ---------------------------------------------------------------------------
extern "C" void kernel_launch(void* const* d_in, const int* in_sizes, int n_in,
                              void* d_out, int out_size, void* d_ws, size_t ws_size,
                              hipStream_t stream) {
  const float* vis = (const float*)d_in[0];   // [16,1024,1024]
  const float* txt = (const float*)d_in[1];   // [16,1024,768]
  const float* Wv  = (const float*)d_in[2];   // [512,1024]
  const float* bv  = (const float*)d_in[3];   // [512]
  const float* Wt  = (const float*)d_in[4];   // [512,768]
  const float* bt  = (const float*)d_in[5];   // [512]
  float* out = (float*)d_out;                 // [16,1024,1024]

  const size_t ROWS = (size_t)16 * SEQ;       // 16384
  // Scratch layout (bf16 staging; all 32B aligned):
  unsigned short* Vbf  = (unsigned short*)d_ws;          // 16384*512   (16 MB)
  unsigned short* Tbf  = Vbf  + ROWS * HDIM;             // 16384*512   (16 MB)
  unsigned short* Xvbf = Tbf  + ROWS * HDIM;             // 16384*1024  (32 MB)
  unsigned short* Xtbf = Xvbf + ROWS * 1024;             // 16384*768   (24 MB)
  unsigned short* Wvbf = Xtbf + ROWS * 768;              // 512*1024    (1 MB)
  unsigned short* Wtbf = Wvbf + (size_t)HDIM * 1024;     // 512*768     (0.75 MB)
  float* norms = (float*)(Wtbf + (size_t)HDIM * 768);    // 32768 f32

  // (1) f32 -> bf16 staging (memory-bound; stays L2-resident).
  cvt_kernel<<<(ROWS * 1024) / 2048, 256, 0, stream>>>(vis, Xvbf);
  cvt_kernel<<<(ROWS * 768)  / 2048, 256, 0, stream>>>(txt, Xtbf);
  cvt_kernel<<<((size_t)HDIM * 1024) / 2048, 256, 0, stream>>>(Wv, Wvbf);
  cvt_kernel<<<((size_t)HDIM * 768)  / 2048, 256, 0, stream>>>(Wt, Wtbf);

  // (2) Projections: block tile 128x128 over [16384 x 512].
  {
    dim3 g(ROWS / 128, HDIM / 128);
    proj_kernel<<<g, 256, 0, stream>>>(Xvbf, Wvbf, bv, Vbf, 1024);
    proj_kernel<<<g, 256, 0, stream>>>(Xtbf, Wtbf, bt, Tbf, 768);
  }

  // (3) Norms over the two contiguous bf16 tensors (32768 rows, 8/block).
  norm_kernel<<<4096, 256, 0, stream>>>(Vbf, norms);

  // (4) Batched cosine-similarity GEMM: grid (N-tiles, M-tiles, batch).
  dim3 g(SEQ / 128, SEQ / 128, 16);
  dots_kernel<<<g, 256, 0, stream>>>(Vbf, Tbf, norms, norms + ROWS, out);
}
